// ScatteringModule_71356586656195
// MI455X (gfx1250) — matrix operand order repaired and checked by reading, not compiled, not run on hardware
//
#include <hip/hip_runtime.h>
#include <cstdint>
#include <cstddef>

// ---------------------------------------------------------------------------
// Problem constants (match reference)
// ---------------------------------------------------------------------------
#define MSZ    32768
#define NSZ    4096
#define WID    128
#define NDEPTH 4
#define KDIM   4096   // K of the big GEMMs == NSZ

typedef __attribute__((ext_vector_type(16))) __bf16 v16bf;
typedef __attribute__((ext_vector_type(8)))  __bf16 v8bf;
typedef __attribute__((ext_vector_type(4)))  __bf16 v4bf;
typedef __attribute__((ext_vector_type(8)))  float  v8f;

// CDNA5 async global->LDS path (device pass only; host pass / older toolchains
// fall back to the register-staged pipeline).
#if defined(__HIP_DEVICE_COMPILE__) && defined(__gfx1250__) && \
    __has_builtin(__builtin_amdgcn_global_load_async_to_lds_b128) && \
    __has_builtin(__builtin_amdgcn_s_wait_asynccnt)
#define USE_ASYNC_LDS 1
// exact parameter pointee type per hipcc diagnostic:
//   "__attribute__((__vector_size__(4 * sizeof(int)))) int *"
typedef int v4i_vs __attribute__((vector_size(4 * sizeof(int))));
#define ASYNC_B128(dst_lds, src_glb)                                          \
    __builtin_amdgcn_global_load_async_to_lds_b128(                           \
        (__attribute__((address_space(1))) v4i_vs*)(src_glb),                 \
        (__attribute__((address_space(3))) v4i_vs*)(dst_lds), 0, 0)
#else
#define USE_ASYNC_LDS 0
#endif

#define BPAD 40   // 32 data + 8 pad halfwords per LDS B-tile row (bank-conflict free)

// 8 column tiles of WMMA against one LDS B tile; fragments preloaded in groups
// of 4 distinct registers so ds_load_b128s batch ahead of the wmma chain.
__device__ __forceinline__ void compute_tile(const __bf16* __restrict__ lbb,
                                             int lm, int hi,
                                             const v16bf& a, v8f acc[8])
{
    #pragma unroll
    for (int jj = 0; jj < 8; jj += 4) {
        v16bf bfr[4];
        #pragma unroll
        for (int u = 0; u < 4; ++u)
            bfr[u] = *(const v16bf*)&lbb[((jj + u) * 16 + lm) * BPAD + hi * 16];
        #pragma unroll
        for (int u = 0; u < 4; ++u)
            acc[jj + u] = __builtin_amdgcn_wmma_f32_16x16x32_bf16(
                false, a, false, bfr[u], (short)0, acc[jj + u], false, false);
    }
}

// ---------------------------------------------------------------------------
// GEMM1: C[rows x 128] (bf16 out) = A[rows x 4096] * s[4096 x 128]
//   sT is bf16 transposed s: sT[n][k] so a WMMA B-fragment is a contiguous
//   32B read. 256 threads = 8 waves; 128x128 output tile; wave w owns rows
//   [16w,16w+16) and all 8 column tiles. B tile (128x32 bf16) double-buffered
//   in LDS, one barrier per K-step, staging overlapped with compute.
// ---------------------------------------------------------------------------
template<bool ABF16>
__global__ __launch_bounds__(256) void gemm1_kernel(const void* __restrict__ Aptr,
                                                    const __bf16* __restrict__ sT,
                                                    __bf16* __restrict__ Cout)
{
    constexpr int K     = KDIM;
    constexpr int NSTEP = K / 32;
    __shared__ __bf16 lb[2][128 * BPAD];     // 2 x 10.2 KB

    const int t    = threadIdx.x;
    const int wave = t >> 5;
    const int lane = t & 31;
    const int lm   = lane & 15;
    const int hi   = lane >> 4;              // 0 = lanes 0-15, 1 = lanes 16-31
    const int mw   = blockIdx.x * 128 + wave * 16;

    // staging assignment: thread t copies 32B (16 bf16) of the B tile
    const int srow  = t >> 1;
    const int shalf = (t & 1) * 16;
    const __bf16* sTrow = sT + (size_t)srow * K + shalf;
    __bf16* ld0 = &lb[0][srow * BPAD + shalf];
    __bf16* ld1 = &lb[1][srow * BPAD + shalf];

    const __bf16* Abf  = (const __bf16*)Aptr;
    const float*  Af   = (const float*)Aptr;
    const size_t  arow = (size_t)(mw + lm) * K;   // this lane's A row
    const int     kofs = hi * 16;                 // A frag K-base per lane half

    // --- A fragment loader (contiguous K layout:
    //     lane<16: M=lane, e0..15 = K kb..kb+15, kb=0;
    //     lane>=16: same with kb=16)
    auto load_a = [&](int k0) -> v16bf {
        v16bf a;
        if constexpr (ABF16) {
            const __bf16* pa = Abf + arow + k0 + kofs;
            v8bf lo = *(const v8bf*)pa;
            v8bf hv = *(const v8bf*)(pa + 8);
            #pragma unroll
            for (int i = 0; i < 8; ++i) { a[i] = lo[i]; a[i + 8] = hv[i]; }
            if (k0 + 512 < K) __builtin_prefetch(pa + 512, 0, 3);  // global_prefetch_b8
        } else {
            const float* pa = Af + arow + k0 + kofs;
            float4 l0 = *(const float4*)(pa);
            float4 l1 = *(const float4*)(pa + 4);
            float4 h0 = *(const float4*)(pa + 8);
            float4 h1 = *(const float4*)(pa + 12);
            a[0]=(__bf16)l0.x; a[1]=(__bf16)l0.y; a[2]=(__bf16)l0.z; a[3]=(__bf16)l0.w;
            a[4]=(__bf16)l1.x; a[5]=(__bf16)l1.y; a[6]=(__bf16)l1.z; a[7]=(__bf16)l1.w;
            a[8]=(__bf16)h0.x; a[9]=(__bf16)h0.y; a[10]=(__bf16)h0.z; a[11]=(__bf16)h0.w;
            a[12]=(__bf16)h1.x; a[13]=(__bf16)h1.y; a[14]=(__bf16)h1.z; a[15]=(__bf16)h1.w;
            if (k0 + 256 < K) __builtin_prefetch(pa + 256, 0, 3);
        }
        return a;
    };

    v8f acc[8] = {};

#if USE_ASYNC_LDS
    // Async pipeline: issue tile ks+1 (ASYNCcnt) after the barrier, compute
    // tile ks.  wait_asynccnt(0) + barrier at top delimit arrival of tile ks.
    ASYNC_B128(ld0,     sTrow);
    ASYNC_B128(ld0 + 8, sTrow + 8);
    v16bf a_cur = load_a(0);
    for (int ks = 0; ks < NSTEP; ++ks) {
        __builtin_amdgcn_s_wait_asynccnt(0);      // own tile-ks writes arrived
        __syncthreads();                          // everyone's arrived / readers done
        const int cur = ks & 1;
        v16bf a_next;
        if (ks + 1 < NSTEP) {
            const __bf16* sn = sTrow + (ks + 1) * 32;
            __bf16* ldn = cur ? ld0 : ld1;
            ASYNC_B128(ldn,     sn);
            ASYNC_B128(ldn + 8, sn + 8);
            a_next = load_a((ks + 1) * 32);
        }
        compute_tile(&lb[cur][0], lm, hi, a_cur, acc);
        a_cur = a_next;
    }
#else
    // Register-staged pipeline, one barrier per K-step:
    //   load g(ks+1) -> regs ; barrier ; compute buf[cur] ; store regs -> buf[nxt]
    v8bf r0 = *(const v8bf*)(sTrow);
    v8bf r1 = *(const v8bf*)(sTrow + 8);
    *(v8bf*)(ld0)     = r0;                       // commit tile 0
    *(v8bf*)(ld0 + 8) = r1;
    v16bf a_cur = load_a(0);
    for (int ks = 0; ks < NSTEP; ++ks) {
        const int cur = ks & 1;
        v16bf a_next;
        if (ks + 1 < NSTEP) {
            const __bf16* sn = sTrow + (ks + 1) * 32;
            r0 = *(const v8bf*)(sn);
            r1 = *(const v8bf*)(sn + 8);
            a_next = load_a((ks + 1) * 32);
        }
        __syncthreads();                          // tile ks stores visible
        compute_tile(&lb[cur][0], lm, hi, a_cur, acc);
        if (ks + 1 < NSTEP) {
            __bf16* ldn = cur ? ld0 : ld1;
            *(v8bf*)(ldn)     = r0;
            *(v8bf*)(ldn + 8) = r1;
        }
        a_cur = a_next;
    }
#endif

    // --- epilogue: C/D layout: VGPR r, lane -> row = r + 8*hi, col = lm
    const size_t rbase = (size_t)(mw + hi * 8);
    #pragma unroll
    for (int j = 0; j < 8; ++j) {
        const int col = j * 16 + lm;
        #pragma unroll
        for (int r = 0; r < 8; ++r)
            Cout[(rbase + r) * WID + col] = (__bf16)acc[j][r];
    }
}

// ---------------------------------------------------------------------------
// GEMM2 (fused): Out[m][i] += tanh( sum_k A[m][k]*W[i][k] + bias[i] )
//   A = msg bf16 [rows x 128]; W row-major [128 x 128] bf16 is exactly the
//   Bt layout (Bt[i][k] = W[i][k]).  W staged once into padded LDS.
// ---------------------------------------------------------------------------
__global__ __launch_bounds__(256) void gemm2_kernel(const __bf16* __restrict__ Abf,
                                                    const __bf16* __restrict__ Wb,
                                                    const float* __restrict__ bias,
                                                    float* __restrict__ Out)
{
    constexpr int K    = WID;      // 128
    constexpr int WPAD = 136;      // 128 data + 8 pad halfwords
    __shared__ __bf16 lw[128 * WPAD];   // 34.8 KB

    const int t    = threadIdx.x;
    const int wave = t >> 5;
    const int lane = t & 31;
    const int lm   = lane & 15;
    const int hi   = lane >> 4;
    const int mw   = blockIdx.x * 128 + wave * 16;

    // stage full W into LDS: thread t copies half a row (64 bf16 = 128 B)
    {
        const int row  = t >> 1;
        const int half = (t & 1) * 64;
        const __bf16* src = Wb + row * K + half;
        __bf16*       dst = &lw[row * WPAD + half];
        #pragma unroll
        for (int i = 0; i < 64; i += 8)
            *(v8bf*)(dst + i) = *(const v8bf*)(src + i);
    }
    __syncthreads();

    v8f acc[8] = {};
    const size_t arow = (size_t)(mw + lm) * K;
    #pragma unroll
    for (int k0 = 0; k0 < K; k0 += 32) {
        const __bf16* pa = Abf + arow + k0 + hi * 16;
        v8bf lo = *(const v8bf*)pa;
        v8bf hv = *(const v8bf*)(pa + 8);
        v16bf a;
        #pragma unroll
        for (int i = 0; i < 8; ++i) { a[i] = lo[i]; a[i + 8] = hv[i]; }
        #pragma unroll
        for (int jj = 0; jj < 8; jj += 4) {
            v16bf bfr[4];
            #pragma unroll
            for (int u = 0; u < 4; ++u)
                bfr[u] = *(const v16bf*)&lw[((jj + u) * 16 + lm) * WPAD + k0 + hi * 16];
            #pragma unroll
            for (int u = 0; u < 4; ++u)
                acc[jj + u] = __builtin_amdgcn_wmma_f32_16x16x32_bf16(
                    false, a, false, bfr[u], (short)0, acc[jj + u], false, false);
        }
    }

    const size_t rbase = (size_t)(mw + hi * 8);
    #pragma unroll
    for (int j = 0; j < 8; ++j) {
        const int col = j * 16 + lm;
        const float bj = bias[col];
        #pragma unroll
        for (int r = 0; r < 8; ++r) {
            const size_t idx = (rbase + r) * WID + col;
            Out[idx] += tanhf(acc[j][r] + bj);
        }
    }
}

// ---------------------------------------------------------------------------
// Elementwise helpers
// ---------------------------------------------------------------------------
__global__ void cvt_bf16_kernel(const float* __restrict__ src,
                                __bf16* __restrict__ dst, size_t n4)
{
    size_t i = (size_t)blockIdx.x * blockDim.x + threadIdx.x;
    const size_t stride = (size_t)gridDim.x * blockDim.x;
    for (; i < n4; i += stride) {
        float4 v = ((const float4*)src)[i];
        v4bf o;
        o[0] = (__bf16)v.x; o[1] = (__bf16)v.y; o[2] = (__bf16)v.z; o[3] = (__bf16)v.w;
        ((v4bf*)dst)[i] = o;
    }
}

__global__ void copy_f32_kernel(const float* __restrict__ src,
                                float* __restrict__ dst, size_t n4)
{
    size_t i = (size_t)blockIdx.x * blockDim.x + threadIdx.x;
    const size_t stride = (size_t)gridDim.x * blockDim.x;
    for (; i < n4; i += stride)
        ((float4*)dst)[i] = ((const float4*)src)[i];
}

// sT[c][r] = (bf16) s[r][c];  s is [rows x 128] f32, sT is [128 x rows] bf16.
__global__ void transpose_cvt_kernel(const float* __restrict__ s,
                                     __bf16* __restrict__ sT, int rows)
{
    const int idx = blockIdx.x * blockDim.x + threadIdx.x;
    if (idx < rows * WID) {
        const int c = idx / rows;
        const int r = idx - c * rows;
        sT[idx] = (__bf16)s[(size_t)r * WID + c];
    }
}

// ---------------------------------------------------------------------------
// Host launcher
// ---------------------------------------------------------------------------
extern "C" void kernel_launch(void* const* d_in, const int* in_sizes, int n_in,
                              void* d_out, int out_size, void* d_ws, size_t ws_size,
                              hipStream_t stream)
{
    (void)in_sizes; (void)n_in; (void)out_size;

    const float* act   = (const float*)d_in[0];
    const float* sact  = (const float*)d_in[1];
    const float* kern  = (const float*)d_in[2];
    const float* skern = (const float*)d_in[3];
    const float* wts   = (const float*)d_in[4];
    const float* bias  = (const float*)d_in[5];

    float* out_a = (float*)d_out;                       // [M x 128]
    float* out_s = out_a + (size_t)MSZ * WID;           // [N x 128]

    // carve workspace (256B aligned slots); big bf16 matrices last so we can
    // fall back to streaming f32 A if ws is too small.
    uintptr_t p = (uintptr_t)d_ws;
    auto take = [&](size_t bytes) -> void* {
        void* r = (void*)p;
        p += (bytes + 255) & ~(size_t)255;
        return r;
    };
    __bf16* wbf  = (__bf16*)take((size_t)NDEPTH * WID * WID * 2);
    __bf16* sT   = (__bf16*)take((size_t)WID * NSZ * 2);
    __bf16* msg  = (__bf16*)take((size_t)MSZ * WID * 2);
    __bf16* smsg = (__bf16*)take((size_t)NSZ * WID * 2);
    __bf16* kbf  = (__bf16*)take((size_t)MSZ * NSZ * 2);
    __bf16* skbf = (__bf16*)take((size_t)NSZ * NSZ * 2);
    const bool conv = (p - (uintptr_t)d_ws) <= ws_size;

    auto cvt = [&](const float* s, __bf16* d, size_t n) {
        const size_t n4 = n / 4;
        int blocks = (int)((n4 + 255) / 256);
        cvt_bf16_kernel<<<blocks, 256, 0, stream>>>(s, d, n4);
    };

    // one-time conversions + output init
    if (conv) {
        cvt(kern,  kbf,  (size_t)MSZ * NSZ);
        cvt(skern, skbf, (size_t)NSZ * NSZ);
    }
    cvt(wts, wbf, (size_t)NDEPTH * WID * WID);
    {
        const size_t na4 = (size_t)MSZ * WID / 4;
        const size_t ns4 = (size_t)NSZ * WID / 4;
        copy_f32_kernel<<<(int)((na4 + 255) / 256), 256, 0, stream>>>(act,  out_a, na4);
        copy_f32_kernel<<<(int)((ns4 + 255) / 256), 256, 0, stream>>>(sact, out_s, ns4);
    }

    for (int d = 0; d < NDEPTH; ++d) {
        // s (f32, current) -> sT bf16 [128 x N]
        transpose_cvt_kernel<<<(NSZ * WID + 255) / 256, 256, 0, stream>>>(out_s, sT, NSZ);

        // msg = kernel @ s ; self_msg = self_kernel @ s
        if (conv) {
            gemm1_kernel<true><<<MSZ / 128, 256, 0, stream>>>(kbf,  sT, msg);
            gemm1_kernel<true><<<NSZ / 128, 256, 0, stream>>>(skbf, sT, smsg);
        } else {
            gemm1_kernel<false><<<MSZ / 128, 256, 0, stream>>>(kern,  sT, msg);
            gemm1_kernel<false><<<NSZ / 128, 256, 0, stream>>>(skern, sT, smsg);
        }

        // a += tanh(msg @ W.T + b) ; s += tanh(self_msg @ W.T + b)
        const __bf16* Wd = wbf + (size_t)d * WID * WID;
        const float*  bd = bias + (size_t)d * WID;
        gemm2_kernel<<<MSZ / 128, 256, 0, stream>>>(msg,  Wd, bd, out_a);
        gemm2_kernel<<<NSZ / 128, 256, 0, stream>>>(smsg, Wd, bd, out_s);
    }
}
